// LSTMLM_30571577213274
// MI455X (gfx1250) — matrix operand order repaired and checked by reading, not compiled, not run on hardware
//
#include <hip/hip_runtime.h>
#include <hip/hip_bf16.h>
#include <math.h>

// ---------------- problem constants ----------------
#define V_ 32000
#define E_ 256
#define H_ 512
#define B_ 16
#define S_ 256
#define G_ 2048          // 4*H
#define M_ 4096          // B*S

typedef __bf16 bf16;
typedef __attribute__((ext_vector_type(16))) __bf16 v16bf;
typedef __attribute__((ext_vector_type(8)))  __bf16 bf16x8;
typedef __attribute__((ext_vector_type(8)))  float  v8f;

// workspace byte offsets (all 256-aligned)
static const size_t OFF_WIHT = 0;                                   // 2048*256 bf16  = 1,048,576 B
static const size_t OFF_FCW  = OFF_WIHT + (size_t)G_ * E_ * 2;      // 32000*512 bf16 = 32,768,000 B
static const size_t OFF_XG   = OFF_FCW  + (size_t)V_ * H_ * 2;      // 4096*2048 f32  = 33,554,432 B
static const size_t OFF_HALL = OFF_XG   + (size_t)M_ * G_ * 4;      // 4096*512 bf16  = 4,194,304 B
static const size_t OFF_HCUR = OFF_HALL + (size_t)M_ * H_ * 2;      // 2 * 16*512 bf16 = 32,768 B
static const size_t OFF_BAR  = OFF_HCUR + (size_t)2 * B_ * H_ * 2;  // barrier uints

// ---------------- WMMA fragment loaders ----------------
// 16-bit A 16x32 (M x K): lane<16 -> elems 0..7 = K[k0+0..7],   8..15 = K[k0+16..23]
//                         lane>=16-> elems 0..7 = K[k0+8..15],  8..15 = K[k0+24..31]
__device__ __forceinline__ v16bf load_a_bf16(const bf16* rowptr, int k0, int khalf) {
  bf16x8 lo = *(const bf16x8*)(rowptr + k0 + khalf);
  bf16x8 hi = *(const bf16x8*)(rowptr + k0 + 16 + khalf);
  v16bf a;
#pragma unroll
  for (int j = 0; j < 8; ++j) { a[j] = lo[j]; a[8 + j] = hi[j]; }
  return a;
}

__device__ __forceinline__ v16bf load_a_f32(const float* rowptr, int k0, int khalf) {
  v16bf a;
#pragma unroll
  for (int j = 0; j < 8; ++j) {
    a[j]     = (bf16)rowptr[k0 + khalf + j];
    a[8 + j] = (bf16)rowptr[k0 + 16 + khalf + j];
  }
  return a;
}

// 16-bit B 32x16 (K x N): lane<16 -> col=lane, elems 0..7 = K[k0+0..7],   8..15 = K[k0+16..23]
//                         lane>=16-> col=lane-16, elems 0..7 = K[k0+8..15], 8..15 = K[k0+24..31]
__device__ __forceinline__ v16bf load_b_bf16(const bf16* colptr, int k0, int kbase) {
  bf16x8 lo = *(const bf16x8*)(colptr + k0 + (kbase >> 1));
  bf16x8 hi = *(const bf16x8*)(colptr + k0 + 16 + (kbase >> 1));
  v16bf b;
#pragma unroll
  for (int j = 0; j < 8; ++j) { b[j] = lo[j]; b[8 + j] = hi[j]; }
  return b;
}

#define WMMA_BF16(a, b, c) \
  __builtin_amdgcn_wmma_f32_16x16x32_bf16(false, (a), false, (b), (short)0, (c), false, false)

__device__ __forceinline__ float sigmoidf_(float x) { return 1.0f / (1.0f + __expf(-x)); }

// CDNA5 async DMA: global -> LDS, 16 bytes per lane, tracked by ASYNCcnt.
__device__ __forceinline__ void async_copy_b128(unsigned lds_byte_off,
                                                unsigned long long gaddr) {
  asm volatile("global_load_async_to_lds_b128 %0, %1, off"
               :: "v"(lds_byte_off), "v"(gaddr) : "memory");
}
__device__ __forceinline__ void async_wait0() {
  asm volatile("s_wait_asynccnt 0x0" ::: "memory");
}

// ---------------- setup kernels ----------------
__global__ void k_init(unsigned int* p) {
  // zero h double-buffer (8192 uints) + barrier pad (16 uints)
  for (int i = threadIdx.x; i < 8208; i += 256) p[i] = 0u;
}

__global__ void k_wihT(const float* __restrict__ wih, bf16* __restrict__ wihT) {
  int idx = blockIdx.x * 256 + threadIdx.x;        // over G_*E_
  int g = idx >> 8, e = idx & 255;
  wihT[idx] = (bf16)wih[(size_t)e * G_ + g];
}

__global__ void k_fcw(const float* __restrict__ fcw, bf16* __restrict__ o) {
  size_t idx = (size_t)blockIdx.x * 256 + threadIdx.x;
  const size_t total = (size_t)V_ * H_;
  for (; idx < total; idx += (size_t)4096 * 256) o[idx] = (bf16)fcw[idx];
}

// ---------------- xg = emb[x] @ W_ih + b_ih + b_hh ----------------
// grid (256, 32), block 128 (4 waves). Row r = s*16 + b. xg layout [S*16][G].
__global__ void k_xg(const int* __restrict__ x, const float* __restrict__ emb,
                     const bf16* __restrict__ wihT, const float* __restrict__ bih,
                     const float* __restrict__ bhh, float* __restrict__ xg) {
  int lane = threadIdx.x & 31, wave = threadIdx.x >> 5;
  int m0 = blockIdx.x * 16;
  int n0 = blockIdx.y * 64 + wave * 16;
  int row = m0 + (lane & 15);
  int s = row >> 4, b = row & 15;
  int tok = x[b * S_ + s];
  const float* arow = emb + (size_t)tok * E_;
  int khalf = (lane < 16) ? 0 : 8;
  int kbase = (lane < 16) ? 0 : 16;
  const bf16* bcol = wihT + (size_t)(n0 + (lane & 15)) * E_;
  v8f acc = {};
#pragma unroll
  for (int kt = 0; kt < E_; kt += 32) {
    v16bf a  = load_a_f32(arow, kt, khalf);
    v16bf bm = load_b_bf16(bcol, kt, kbase);
    acc = WMMA_BF16(a, bm, acc);
  }
  int n = n0 + (lane & 15);
  float bias = bih[n] + bhh[n];
  int mhi = (lane >> 4) * 8;
#pragma unroll
  for (int i = 0; i < 8; ++i)
    xg[(size_t)(m0 + mhi + i) * G_ + n] = acc[i] + bias;
}

// ---------------- LSTM recurrence: 16 persistent WGs, W_hh slice in LDS ----------------
// WG k owns h columns [k*32, k*32+32) => gate cols {q*512 + k*32 + j : q<4, j<32}.
// LDS: Bslice bf16[128][512] | hb bf16[16][512] | gates f32[16][128] | c f32[16][32]
__global__ void k_lstm(const float* __restrict__ whh, const float* __restrict__ xg,
                       bf16* __restrict__ hcur, bf16* __restrict__ hall,
                       float* __restrict__ hT, float* __restrict__ cT,
                       unsigned int* __restrict__ bar) {
  extern __shared__ char smem[];
  bf16*  bsl = (bf16*)smem;                  // 131072 B
  bf16*  hb  = (bf16*)(smem + 131072);       // 16384 B
  float* gt  = (float*)(smem + 147456);      // 8192 B
  float* cs  = (float*)(smem + 155648);      // 2048 B
  const int k = blockIdx.x, tid = threadIdx.x;
  const int lane = tid & 31, wave = tid >> 5;

  // one-time: load + transpose + convert W_hh slice into LDS
  for (int idx = tid; idx < 128 * 512; idx += 256) {
    int c = idx >> 9, kk = idx & 511;
    int gc = (c >> 5) * 512 + k * 32 + (c & 31);
    bsl[idx] = (bf16)whh[(size_t)kk * G_ + gc];
  }
  for (int idx = tid; idx < 512; idx += 256) cs[idx] = 0.0f;
  __syncthreads();

  const int khalf = (lane < 16) ? 0 : 8;
  const int kbase = (lane < 16) ? 0 : 16;
  const int mhi   = (lane >> 4) * 8;
  const int n_local = wave * 16 + (lane & 15);
  const int gc = (n_local >> 5) * 512 + k * 32 + (n_local & 31);
  const bf16* bcol = bsl + (size_t)n_local * H_;
  const bf16* arow = hb + (size_t)(lane & 15) * H_;
  const unsigned hb_lds = (unsigned)(unsigned long long)(void*)hb + (unsigned)tid * 16u;

  for (int t = 0; t < S_; ++t) {
    // pull full h_t (double-buffered in L2) into LDS via async DMA (no VGPR staging)
    {
      unsigned long long gsrc =
          (unsigned long long)(const void*)(hcur + (size_t)(t & 1) * (B_ * H_)) +
          (unsigned long long)tid * 16ull;
#pragma unroll
      for (int it = 0; it < 4; ++it)
        async_copy_b128(hb_lds + (unsigned)it * 4096u, gsrc + (unsigned long long)it * 4096ull);
      async_wait0();
    }
    __syncthreads();

    // acc initialized from xg contribution (acts as C matrix)
    v8f acc;
    const float* xgp = xg + (size_t)t * 16 * G_ + gc;
#pragma unroll
    for (int i = 0; i < 8; ++i) acc[i] = xgp[(size_t)(mhi + i) * G_];

#pragma unroll
    for (int kt = 0; kt < H_; kt += 32) {
      v16bf a  = load_a_bf16(arow, kt, khalf);
      v16bf bm = load_b_bf16(bcol, kt, kbase);
      acc = WMMA_BF16(a, bm, acc);
    }
#pragma unroll
    for (int i = 0; i < 8; ++i) gt[(mhi + i) * 128 + n_local] = acc[i];
    __syncthreads();

    // pointwise cell update for owned 16x32 h/c elements
    for (int e = tid; e < 512; e += 256) {
      int b = e >> 5, j = e & 31;
      float it = sigmoidf_(gt[b * 128 + j]);
      float ft = sigmoidf_(gt[b * 128 + 32 + j]);
      float gg = tanhf(gt[b * 128 + 64 + j]);
      float ot = sigmoidf_(gt[b * 128 + 96 + j]);
      float cn = ft * cs[e] + it * gg;
      float hn = ot * tanhf(cn);
      cs[e] = cn;
      int col = k * 32 + j;
      hcur[(size_t)((t + 1) & 1) * (B_ * H_) + b * H_ + col] = (bf16)hn;
      hall[((size_t)b * S_ + t) * H_ + col] = (bf16)hn;
      if (t == S_ - 1) { hT[b * H_ + col] = hn; cT[b * H_ + col] = cn; }
    }
    __syncthreads();

    // device-wide barrier across the 16 WGs (monotonic counter, zeroed by k_init)
    if (tid == 0) {
      __threadfence();
      __hip_atomic_fetch_add(bar, 1u, __ATOMIC_RELEASE, __HIP_MEMORY_SCOPE_AGENT);
      unsigned int target = (unsigned int)(16 * (t + 1));
      while (__hip_atomic_load(bar, __ATOMIC_ACQUIRE, __HIP_MEMORY_SCOPE_AGENT) < target) {}
    }
    __syncthreads();
  }
}

// ---------------- logits = h_all @ fc_w^T + fc_b ----------------
// grid (32, 250), block 256 = 8 waves stacked over M; WG tile 128M x 128N.
// Each wave: 16M x 128N (8 N-tiles, 8 accumulators, A fragment reused 8x;
// B rows shared by all 8 waves through WGP$).
__global__ void k_logits(const bf16* __restrict__ hall, const bf16* __restrict__ fcw,
                         const float* __restrict__ fcb, float* __restrict__ out) {
  int lane = threadIdx.x & 31, wave = threadIdx.x >> 5;
  int m0 = blockIdx.x * 128 + wave * 16;
  int n0 = blockIdx.y * 128;
  int khalf = (lane < 16) ? 0 : 8;
  int kbase = (lane < 16) ? 0 : 16;
  const bf16* arow = hall + (size_t)(m0 + (lane & 15)) * H_;
  const bf16* b0   = fcw + (size_t)(n0 + (lane & 15)) * H_;
  __builtin_prefetch(arow, 0, 1);
  __builtin_prefetch(b0, 0, 1);
  v8f acc[8] = {v8f{}, v8f{}, v8f{}, v8f{}, v8f{}, v8f{}, v8f{}, v8f{}};
  for (int kt = 0; kt < H_; kt += 32) {
    v16bf a = load_a_bf16(arow, kt, khalf);
#pragma unroll
    for (int nt = 0; nt < 8; ++nt) {
      v16bf bm = load_b_bf16(b0 + (size_t)nt * 16 * H_, kt, kbase);
      acc[nt] = WMMA_BF16(a, bm, acc[nt]);
    }
  }
  int mhi = (lane >> 4) * 8;
#pragma unroll
  for (int nt = 0; nt < 8; ++nt) {
    int n = n0 + nt * 16 + (lane & 15);
    float bias = fcb[n];
#pragma unroll
    for (int i = 0; i < 8; ++i)
      out[(size_t)(m0 + mhi + i) * V_ + n] = acc[nt][i] + bias;
  }
}

// ---------------- in-place log_softmax per row ----------------
__global__ void k_lsm(float* __restrict__ out) {
  __shared__ float red[1024];
  const size_t row = blockIdx.x;
  float* p = out + row * V_;
  int tid = threadIdx.x;
  float m = -3.4e38f;
  for (int i = tid; i < V_; i += 1024) m = fmaxf(m, p[i]);
  red[tid] = m; __syncthreads();
  for (int s = 512; s > 0; s >>= 1) {
    if (tid < s) red[tid] = fmaxf(red[tid], red[tid + s]);
    __syncthreads();
  }
  m = red[0]; __syncthreads();
  float sum = 0.0f;
  for (int i = tid; i < V_; i += 1024) sum += __expf(p[i] - m);
  red[tid] = sum; __syncthreads();
  for (int s = 512; s > 0; s >>= 1) {
    if (tid < s) red[tid] += red[tid + s];
    __syncthreads();
  }
  float lse = m + __logf(red[0]);
  for (int i = tid; i < V_; i += 1024) p[i] = p[i] - lse;
}

// ---------------- host launcher ----------------
extern "C" void kernel_launch(void* const* d_in, const int* in_sizes, int n_in,
                              void* d_out, int out_size, void* d_ws, size_t ws_size,
                              hipStream_t stream) {
  (void)in_sizes; (void)n_in; (void)out_size; (void)ws_size;
  const int*   x   = (const int*)d_in[0];
  const float* emb = (const float*)d_in[1];
  const float* wih = (const float*)d_in[2];
  const float* whh = (const float*)d_in[3];
  const float* bih = (const float*)d_in[4];
  const float* bhh = (const float*)d_in[5];
  const float* fcw = (const float*)d_in[6];
  const float* fcb = (const float*)d_in[7];
  float* out = (float*)d_out;
  char*  ws  = (char*)d_ws;

  bf16*  wihT = (bf16*)(ws + OFF_WIHT);
  bf16*  fcwb = (bf16*)(ws + OFF_FCW);
  float* xg   = (float*)(ws + OFF_XG);
  bf16*  hall = (bf16*)(ws + OFF_HALL);
  bf16*  hcur = (bf16*)(ws + OFF_HCUR);
  unsigned int* bar = (unsigned int*)(ws + OFF_BAR);
  float* hT = out + (size_t)M_ * V_;
  float* cT = hT + (size_t)B_ * H_;

  k_init<<<1, 256, 0, stream>>>((unsigned int*)(ws + OFF_HCUR));
  k_wihT<<<(G_ * E_) / 256, 256, 0, stream>>>(wih, wihT);
  k_fcw<<<4096, 256, 0, stream>>>(fcw, fcwb);
  k_xg<<<dim3(M_ / 16, G_ / 64), 128, 0, stream>>>(x, emb, wihT, bih, bhh, xg);
  k_lstm<<<16, 256, 157696, stream>>>(whh, xg, hcur, hall, hT, cT, bar);
  k_logits<<<dim3(M_ / 128, V_ / 128), 256, 0, stream>>>(hall, fcwb, fcb, out);
  k_lsm<<<M_, 1024, 0, stream>>>(out);
}
